// MyModel_61933428416173
// MI455X (gfx1250) — compile-verified, matrix operand order checked by reading, not run on hardware
//
#include <hip/hip_runtime.h>

// Per-row mode of a [N,64] fp32 matrix with values in {0..7}.
// histogram = E(16x512) x S(512x8) via 8 accumulated V_WMMA_I32_16X16X64_IU8
// per 16-row tile; S[j][L] = ((j%8)==L) is constant. Each element one-hot
// expands to u64 = 1 << 8v in an 8-aligned expanded-K slot; slots assigned so
// each lane reads 4 CONTIGUOUS elements per chunk (one b128 load).
// sched_barriers pin the pipeline: loads -> all A builds (64 distinct VGPRs)
// -> 8 back-to-back WMMAs (alternating accumulators, no hazard NOPs) -> tail.
// Reduction uses immediate-pattern ds_swizzle (lane ^ m), cheaper than
// shfl_xor's ds_bpermute + address math.

typedef int   v8i __attribute__((ext_vector_type(8)));
typedef float v4f __attribute__((ext_vector_type(4)));

__device__ __forceinline__ int rmax8(int k) {
    // group-of-32 swizzle: new_lane = ((lane & 0x1f) | 0) ^ m ; masks 1,2,4
    // stay within each 8-lane group, so this matches shfl_xor width 8 here.
    k = max(k, __builtin_amdgcn_ds_swizzle(k, 0x041F));  // xor 1
    k = max(k, __builtin_amdgcn_ds_swizzle(k, 0x081F));  // xor 2
    k = max(k, __builtin_amdgcn_ds_swizzle(k, 0x101F));  // xor 4
    return k;
}

__global__ __launch_bounds__(256) void mode_rows_wmma(const float* __restrict__ x,
                                                      float* __restrict__ out,
                                                      int nrows) {
    const int lane = threadIdx.x & 31;
    const int wave = threadIdx.x >> 5;
    const int tile = blockIdx.x * 8 + wave;       // one 16-row tile per wave
    const int tileBase = tile * 16;
    if (tileBase >= nrows) return;                // wave-uniform

    const int m    = lane & 15;                   // row within tile
    const int half = lane >> 4;                   // element halves 0-31 / 32-63

    // Clamp row for (never-taken with N=1M) partial tiles; keeps EXEC all-1s.
    const int rowIdx = min(tileBase + m, nrows - 1);
    const v4f* xr4 = (const v4f*)(x + (size_t)rowIdx * 64 + half * 32);

    // Constant selector B: B[k][n] = ((k & 7) == n) ? 1 : 0 for n < 8.
    // Byte (VGPR r, byte p) of lane n maps to K == (4r + p) (mod 8) in all
    // documented 64x16 iu8 B layouts.
    v8i B;
    {
        const int n16 = lane & 15;
        const unsigned int bsel = (n16 < 8) ? (1u << ((n16 & 3) * 8)) : 0u;
        const int rpar = (n16 >> 2) & 1;
        #pragma unroll
        for (int r = 0; r < 8; ++r)
            B[r] = (int)(((r & 1) == rpar) ? bsel : 0u);
    }

    // Phase 1: issue all streaming loads (read-once data, NT hint).
    v4f f[8];
    #pragma unroll
    for (int c = 0; c < 8; ++c)
        f[c] = __builtin_nontemporal_load(&xr4[c]);   // elems half*32+4c..+3

    // Phase 2: build all 8 A matrices; must finish before any WMMA issues,
    // forcing 8 disjoint live A register sets.
    v8i A[8];
    #pragma unroll
    for (int c = 0; c < 8; ++c) {
        #pragma unroll
        for (int j = 0; j < 4; ++j) {
            // pair j <- one-hot of element (half*32 + 4c + j)
            int sh = (int)(f[c][j] * 8.0f);           // 8*v, v in 0..7
            unsigned long long oh = 1ull << sh;       // 8-byte one-hot
            A[c][2 * j]     = (int)(unsigned int)oh;
            A[c][2 * j + 1] = (int)(unsigned int)(oh >> 32);
        }
    }

    __builtin_amdgcn_sched_barrier(0);   // no A-build may sink past a WMMA

    // Phase 3: 8 back-to-back WMMAs; alternating accumulators, disjoint A/D
    // registers -> no inter-WMMA hazard NOPs.
    v8i acc0 = {};
    v8i acc1 = {};
    #pragma unroll
    for (int c = 0; c < 8; c += 2) {
        acc0 = __builtin_amdgcn_wmma_i32_16x16x64_iu8(false, A[c], false, B,
                                                      acc0, false, false);
        acc1 = __builtin_amdgcn_wmma_i32_16x16x64_iu8(false, A[c + 1], false, B,
                                                      acc1, false, false);
    }

    __builtin_amdgcn_sched_barrier(0);   // keep tail out of the WMMA chain

    // D layout: VGPR j, lanes 0-15 -> row tileBase+j  (level = lane, 0..7),
    //                    lanes 16-31 -> row tileBase+8+j (level = lane-16).
    // key = (count<<3) | (7-L): max -> highest count, ties -> smallest level.
    const int inv7 = 7 - (lane & 7);
    int k[8];
    #pragma unroll
    for (int j = 0; j < 8; ++j) {
        int cnt = acc0[j] + acc1[j];
        // Lanes 8-15 / 24-31 hold garbage but their xor-8 groups are never read.
        k[j] = rmax8((cnt << 3) | inv7);
    }

    // Lane l (l&15 < 8) picks row (l&7)'s key: 7-cndmask select tree.
    int t0 = (lane & 1) ? k[1] : k[0];
    int t1 = (lane & 1) ? k[3] : k[2];
    int t2 = (lane & 1) ? k[5] : k[4];
    int t3 = (lane & 1) ? k[7] : k[6];
    int u0 = (lane & 2) ? t1 : t0;
    int u1 = (lane & 2) ? t3 : t2;
    int kk = (lane & 4) ? u1 : u0;

    if ((lane & 15) < 8) {
        int r = tileBase + (lane >> 4) * 8 + (lane & 7);
        if (r < nrows)
            out[r] = (float)(7 - (kk & 7));      // decoded mode value
    }
}

extern "C" void kernel_launch(void* const* d_in, const int* in_sizes, int n_in,
                              void* d_out, int out_size, void* d_ws, size_t ws_size,
                              hipStream_t stream) {
    const float* x = (const float*)d_in[0];
    float* out = (float*)d_out;
    const int nrows = out_size;                       // N = 1048576
    const int tiles = (nrows + 15) / 16;              // one wave per 16-row tile
    const int blocks = (tiles + 7) / 8;               // 8 waves per block
    mode_rows_wmma<<<blocks, 256, 0, stream>>>(x, out, nrows);
}
